// Image_Vox_Local_Sampler_89481348644938
// MI455X (gfx1250) — matrix operand order repaired and checked by reading, not compiled, not run on hardware
//
#include <hip/hip_runtime.h>
#include <hip/hip_bf16.h>
#include <cstdint>
#include <cstddef>

// ---------------------------------------------------------------------------
// Types & WMMA helpers (CDNA5 / gfx1250, wave32)
// ---------------------------------------------------------------------------
typedef __attribute__((ext_vector_type(16))) __bf16 v16bf;
typedef __attribute__((ext_vector_type(8)))  float  v8f;

#define DEVI static __device__ __forceinline__

// A-matrix 16x32 bf16 fragment. rowChunk = this lane's row pointer at the start
// of a 32-wide K chunk (64B aligned). Lanes 0-15 hold K=0..15,
// lanes 16-31 hold K=16..31 (same K order as B).
DEVI v16bf load_a_frag(const __bf16* rowChunk, int hi) {
  union { v16bf v; uint4 q[2]; } u;
  const uint4* p = (const uint4*)rowChunk;
  u.q[0] = p[hi ? 2 : 0];
  u.q[1] = p[hi ? 3 : 1];
  return u.v;
}
// B-matrix 32x16 bf16 fragment. colChunk = this lane's column pointer at the
// start of a 32-wide K chunk. Lanes 0-15 hold K=0..15, lanes 16-31 K=16..31.
DEVI v16bf load_b_frag(const __bf16* colChunk, int hi) {
  union { v16bf v; uint4 q[2]; } u;
  const uint4* p = (const uint4*)colChunk;
  u.q[0] = p[hi ? 2 : 0];
  u.q[1] = p[hi ? 3 : 1];
  return u.v;
}
DEVI v8f wmma_bf16(v16bf a, v16bf b, v8f c) {
  return __builtin_amdgcn_wmma_f32_16x16x32_bf16(
      /*neg_a=*/false, a, /*neg_b=*/false, b,
      /*c_mod=*/(short)0, c, /*reuse_a=*/false, /*reuse_b=*/false);
}

// ---------------------------------------------------------------------------
// Problem constants
// ---------------------------------------------------------------------------
#define RESO   32
#define INNER  128
#define IN_CH  1280
#define OUT_CH 64
#define HEADS  8
#define BATCH  4
#define NROWS  (3 * BATCH * RESO * RESO)   // 12288 attention rows
#define GP     34                          // padded voxel dim

// ---------------------------------------------------------------------------
// Prep kernels: layout transposes + f32 -> bf16 conversion
// ---------------------------------------------------------------------------
__global__ void k_cvt_imT(const float* __restrict__ img, __bf16* __restrict__ imT) {
  int idx = blockIdx.x * 256 + threadIdx.x;            // over 4*1280*256
  if (idx >= BATCH * IN_CH * 256) return;
  int b  = idx / (IN_CH * 256);
  int r  = idx - b * (IN_CH * 256);
  int c  = r / 256;
  int hw = r - c * 256;
  imT[((size_t)b * 256 + hw) * IN_CH + c] = (__bf16)img[idx];
}

__global__ void k_cvt_bf16(const float* __restrict__ src, __bf16* __restrict__ dst, int n) {
  int i = blockIdx.x * 256 + threadIdx.x;
  if (i < n) dst[i] = (__bf16)src[i];
}

// vox_w [O=128][C=128][3][3][3] -> wT [tap=27][O][C] (C contiguous)
__global__ void k_cvt_wconv(const float* __restrict__ vw, __bf16* __restrict__ wT) {
  int idx = blockIdx.x * 256 + threadIdx.x;            // over 128*128*27
  if (idx >= 128 * 128 * 27) return;
  int tap = idx % 27;
  int c   = (idx / 27) % 128;
  int o   = idx / (27 * 128);
  wT[((size_t)tap * 128 + o) * 128 + c] = (__bf16)vw[idx];
}

// ---------------------------------------------------------------------------
// Weight folding: Wk' = inW_k @ k_w, Wv' = inW_v @ v_w (bf16), Wq' = inW_q @ q_w (f32)
// ---------------------------------------------------------------------------
__global__ void k_combine_w(const float* __restrict__ inW,
                            const float* __restrict__ kw, const float* __restrict__ vw,
                            const float* __restrict__ qw,
                            __bf16* __restrict__ wkO, __bf16* __restrict__ wvO,
                            float* __restrict__ wqO) {
  int idx = blockIdx.x * 256 + threadIdx.x;            // over 3*128*128
  if (idx >= 3 * 128 * 128) return;
  int m = idx >> 14;
  int r = idx & 16383;
  int o = r >> 7;
  int c = r & 127;
  const float* lin = (m == 0) ? kw : (m == 1) ? vw : qw;
  const float* W   = inW + ((m == 0) ? 128 * 128 : (m == 1) ? 2 * 128 * 128 : 0);
  float s = 0.f;
  #pragma unroll 8
  for (int t = 0; t < 128; ++t) s += W[o * 128 + t] * lin[t * 128 + c];
  if (m == 0)      wkO[o * 128 + c] = (__bf16)s;
  else if (m == 1) wvO[o * 128 + c] = (__bf16)s;
  else             wqO[o * 128 + c] = s;
}

DEVI float pe_val(int r, int t) {
  int i = t >> 1;
  float div = __expf(-(float)(2 * i) * (9.210340371976184f / 128.0f));
  float a = (float)r * div;
  return (t & 1) ? __cosf(a) : __sinf(a);
}

// biasK[r][o] = (k_b + PE[r]) @ Wk^T + inB_k ;  same for V ;  biasQ = q_b @ Wq^T + inB_q
__global__ void k_combine_b(const float* __restrict__ inW, const float* __restrict__ inB,
                            const float* __restrict__ kb, const float* __restrict__ vb,
                            const float* __restrict__ qb,
                            float* __restrict__ bkO, float* __restrict__ bvO,
                            float* __restrict__ bqO) {
  int idx = blockIdx.x * 256 + threadIdx.x;            // 2*32*128 + 128 = 8320
  if (idx < 4096) {
    int r = idx >> 7, o = idx & 127;
    float s = inB[128 + o];
    for (int t = 0; t < 128; ++t) s += (kb[t] + pe_val(r, t)) * inW[(128 + o) * 128 + t];
    bkO[idx] = s;
  } else if (idx < 8192) {
    int j = idx - 4096;
    int r = j >> 7, o = j & 127;
    float s = inB[256 + o];
    for (int t = 0; t < 128; ++t) s += (vb[t] + pe_val(r, t)) * inW[(256 + o) * 128 + t];
    bvO[j] = s;
  } else if (idx < 8320) {
    int o = idx - 8192;
    float s = inB[o];
    for (int t = 0; t < 128; ++t) s += qb[t] * inW[o * 128 + t];
    bqO[o] = s;
  }
}

// ---------------------------------------------------------------------------
// img_proj: [1024 x 1280] @ [1280 x 128] WMMA GEMM -> imf f32 [b][hw][o]
// 32 blocks x 128 threads (4 waves). Wave: 2 M-tiles x 2 N-tiles.
// ---------------------------------------------------------------------------
__global__ __launch_bounds__(128)
void k_imgproj(const __bf16* __restrict__ imT, const __bf16* __restrict__ wimg,
               const float* __restrict__ bias, float* __restrict__ imf) {
  int mBase = blockIdx.x * 32;
  int lane = threadIdx.x & 31, wave = threadIdx.x >> 5;
  int col = lane & 15, hi = lane >> 4;
  v8f acc[2][2] = {};
  const __bf16* arow0 = imT + (size_t)(mBase + col) * IN_CH;
  const __bf16* arow1 = imT + (size_t)(mBase + 16 + col) * IN_CH;
  for (int kc = 0; kc < IN_CH / 32; ++kc) {
    v16bf a0 = load_a_frag(arow0 + kc * 32, hi);
    v16bf a1 = load_a_frag(arow1 + kc * 32, hi);
    #pragma unroll
    for (int nt = 0; nt < 2; ++nt) {
      int o = wave * 32 + nt * 16 + col;
      v16bf bf = load_b_frag(wimg + (size_t)o * IN_CH + kc * 32, hi);
      acc[0][nt] = wmma_bf16(a0, bf, acc[0][nt]);
      acc[1][nt] = wmma_bf16(a1, bf, acc[1][nt]);
    }
  }
  #pragma unroll
  for (int mt = 0; mt < 2; ++mt)
    #pragma unroll
    for (int nt = 0; nt < 2; ++nt) {
      int o = wave * 32 + nt * 16 + col;
      float bo = bias[o];
      #pragma unroll
      for (int v = 0; v < 8; ++v) {
        int m = mBase + mt * 16 + (hi ? v + 8 : v);
        imf[(size_t)m * 128 + o] = acc[mt][nt][v] + bo;
      }
    }
}

// ---------------------------------------------------------------------------
// Sampler: project voxel, bilinear-sample imf, write bf16 into zero-padded
// grid_pad [b][x+1][y+1][z+1][c]. One block per voxel, 128 threads = channels.
// ---------------------------------------------------------------------------
__global__ __launch_bounds__(128)
void k_sampler(const float* __restrict__ imf, const float* __restrict__ pm,
               __bf16* __restrict__ gpad) {
  int blk = blockIdx.x;                                // b*32768 + x*1024 + y*32 + z
  int b = blk >> 15, x = (blk >> 10) & 31, y = (blk >> 5) & 31, z = blk & 31;
  const float s = 2.0f * (1.0f + 0.1f + 1e-5f);
  float vx = ((float)x * (1.0f / 31.0f) - 0.5f) * s;
  float vy = ((float)y * (1.0f / 31.0f) - 0.5f) * s;
  float vz = ((float)z * (1.0f / 31.0f) - 0.5f) * s;
  const float* M = pm + b * 12;
  float c0 = M[0] * vx + M[1] * vy + M[2]  * vz + M[3];
  float c1 = M[4] * vx + M[5] * vy + M[6]  * vz + M[7];
  float c2 = M[8] * vx + M[9] * vy + M[10] * vz + M[11];
  float inv = 1.0f / c2;
  float ix = (c0 * inv) * (15.0f / 223.0f);
  float iy = (c1 * inv) * (15.0f / 223.0f);
  float fx0 = floorf(ix), fy0 = floorf(iy);
  float fx = ix - fx0, fy = iy - fy0;
  int x0 = (int)fx0, y0 = (int)fy0;
  float w00 = (1.f - fx) * (1.f - fy), w10 = fx * (1.f - fy);
  float w01 = (1.f - fx) * fy,         w11 = fx * fy;
  int c = threadIdx.x;
  const float* base = imf + (size_t)b * 256 * 128 + c;
  float acc = 0.f;
  if (x0 >= 0     && x0 < 16     && y0 >= 0     && y0 < 16)     acc += w00 * base[(size_t)(y0 * 16 + x0) * 128];
  if (x0 + 1 >= 0 && x0 + 1 < 16 && y0 >= 0     && y0 < 16)     acc += w10 * base[(size_t)(y0 * 16 + x0 + 1) * 128];
  if (x0 >= 0     && x0 < 16     && y0 + 1 >= 0 && y0 + 1 < 16) acc += w01 * base[(size_t)((y0 + 1) * 16 + x0) * 128];
  if (x0 + 1 >= 0 && x0 + 1 < 16 && y0 + 1 >= 0 && y0 + 1 < 16) acc += w11 * base[(size_t)((y0 + 1) * 16 + x0 + 1) * 128];
  gpad[((((size_t)(b * GP + x + 1)) * GP + (y + 1)) * GP + (z + 1)) * 128 + c] = (__bf16)acc;
}

// ---------------------------------------------------------------------------
// 3x3x3 conv 128->128 as implicit WMMA GEMM over the padded grid.
// One block per (b,x,y) row of 32 z. 4 waves; wave: 2 M-tiles x 2 N-tiles.
// K = 27 taps x 128 channels. Output channel-last bf16 vox[b][x][y][z][o] (+bias).
// ---------------------------------------------------------------------------
__global__ __launch_bounds__(128)
void k_conv3d(const __bf16* __restrict__ gpad, const __bf16* __restrict__ wT,
              const float* __restrict__ voxb, __bf16* __restrict__ vox) {
  int blk = blockIdx.x;                                // b*1024 + x*32 + y
  int b = blk >> 10, x = (blk >> 5) & 31, y = blk & 31;
  int lane = threadIdx.x & 31, wave = threadIdx.x >> 5;
  int col = lane & 15, hi = lane >> 4;
  v8f acc[2][2] = {};
  for (int tap = 0; tap < 27; ++tap) {
    int dx = tap / 9, dy = (tap / 3) % 3, dz = tap % 3;
    size_t rowBase = (((size_t)(b * GP + x + dx) * GP) + (y + dy)) * GP;
    const __bf16* arow0 = gpad + ((rowBase + (col + dz))      << 7);
    const __bf16* arow1 = gpad + ((rowBase + (col + 16 + dz)) << 7);
    const __bf16* wtap  = wT + ((size_t)tap << 14);
    #pragma unroll
    for (int kc = 0; kc < 4; ++kc) {
      v16bf a0 = load_a_frag(arow0 + kc * 32, hi);
      v16bf a1 = load_a_frag(arow1 + kc * 32, hi);
      #pragma unroll
      for (int nt = 0; nt < 2; ++nt) {
        int o = wave * 32 + nt * 16 + col;
        v16bf bf = load_b_frag(wtap + ((size_t)o << 7) + kc * 32, hi);
        acc[0][nt] = wmma_bf16(a0, bf, acc[0][nt]);
        acc[1][nt] = wmma_bf16(a1, bf, acc[1][nt]);
      }
    }
  }
  size_t outBase = ((size_t)((b * 32 + x) * 32 + y)) << 12;   // *32*128
  #pragma unroll
  for (int mt = 0; mt < 2; ++mt)
    #pragma unroll
    for (int nt = 0; nt < 2; ++nt) {
      int o = wave * 32 + nt * 16 + col;
      float bo = voxb[o];
      #pragma unroll
      for (int v = 0; v < 8; ++v) {
        int z = mt * 16 + (hi ? v + 8 : v);
        vox[outBase + ((size_t)z << 7) + o] = (__bf16)(acc[mt][nt][v] + bo);
      }
    }
}

// ---------------------------------------------------------------------------
// Fused per-row: K/V projection (WMMA, folded weights), query projection,
// 8-head SDPA over 32 keys, out_proj. One block per attention row l.
// ---------------------------------------------------------------------------
__global__ __launch_bounds__(128)
void k_kv_attn(const __bf16* __restrict__ vox,
               const __bf16* __restrict__ wk, const __bf16* __restrict__ wv,
               const float* __restrict__ wq,
               const float* __restrict__ bk, const float* __restrict__ bv,
               const float* __restrict__ bq,
               const float* __restrict__ triplane,
               const float* __restrict__ outw, const float* __restrict__ outb,
               __bf16* __restrict__ orow) {
  __shared__ float Ks[32][128];
  __shared__ float Vs[32][128];
  __shared__ float Qs[128];
  __shared__ float qin[128];
  __shared__ float attn[8][32];
  __shared__ float os[128];

  int l = blockIdx.x;
  int p = l >> 12;                 // plane 0:xz 1:xy 2:yz
  int t = l & 4095;
  int b = t >> 10;
  int i = (t >> 5) & 31;
  int j = t & 31;
  int lane = threadIdx.x & 31, wave = threadIdx.x >> 5;
  int col = lane & 15, hi = lane >> 4;

  // Row r of the key/value sequence -> vox element index (channel-last)
  auto rowIdx = [&](int r) -> size_t {
    int X, Y, Z;
    if (p == 0)      { X = j; Y = r; Z = i; }   // rows (b,z,x), seq over y
    else if (p == 1) { X = j; Y = i; Z = r; }   // rows (b,y,x), seq over z
    else             { X = r; Y = j; Z = i; }   // rows (b,z,y), seq over x
    return ((((size_t)b * 32 + X) * 32 + Y) * 32 + Z) << 7;
  };
  const __bf16* arow0 = vox + rowIdx(col);
  const __bf16* arow1 = vox + rowIdx(col + 16);

  v8f ak[2][2] = {}, av[2][2] = {};
  #pragma unroll
  for (int kc = 0; kc < 4; ++kc) {
    v16bf a0 = load_a_frag(arow0 + kc * 32, hi);
    v16bf a1 = load_a_frag(arow1 + kc * 32, hi);
    #pragma unroll
    for (int nt = 0; nt < 2; ++nt) {
      int o = wave * 32 + nt * 16 + col;
      v16bf bkf = load_b_frag(wk + ((size_t)o << 7) + kc * 32, hi);
      v16bf bvf = load_b_frag(wv + ((size_t)o << 7) + kc * 32, hi);
      ak[0][nt] = wmma_bf16(a0, bkf, ak[0][nt]);
      ak[1][nt] = wmma_bf16(a1, bkf, ak[1][nt]);
      av[0][nt] = wmma_bf16(a0, bvf, av[0][nt]);
      av[1][nt] = wmma_bf16(a1, bvf, av[1][nt]);
    }
  }
  #pragma unroll
  for (int mt = 0; mt < 2; ++mt)
    #pragma unroll
    for (int nt = 0; nt < 2; ++nt) {
      int o = wave * 32 + nt * 16 + col;
      #pragma unroll
      for (int v = 0; v < 8; ++v) {
        int r = mt * 16 + (hi ? v + 8 : v);
        Ks[r][o] = ak[mt][nt][v] + bk[r * 128 + o];
        Vs[r][o] = av[mt][nt][v] + bv[r * 128 + o];
      }
    }
  // stage query features (strided channel gather from triplane f32)
  int c = threadIdx.x;
  qin[c] = triplane[(((size_t)b * 128 + c) * 96 + (p * 32 + i)) * 32 + j];
  __syncthreads();

  // folded query projection (VALU: M=1 not worth a WMMA)
  {
    float qv = bq[c];
    #pragma unroll 8
    for (int k = 0; k < 128; ++k) qv += qin[k] * wq[c * 128 + k];
    Qs[c] = qv;
  }
  __syncthreads();

  // scores [8 heads][32 keys], scale = 1/sqrt(16)
  for (int sidx = threadIdx.x; sidx < 256; sidx += 128) {
    int h = sidx >> 5, r = sidx & 31;
    float sc = 0.f;
    #pragma unroll
    for (int d = 0; d < 16; ++d) sc += Qs[h * 16 + d] * Ks[r][h * 16 + d];
    attn[h][r] = sc * 0.25f;
  }
  __syncthreads();
  if (threadIdx.x < 8) {
    int h = threadIdx.x;
    float mx = -1e30f;
    for (int r = 0; r < 32; ++r) mx = fmaxf(mx, attn[h][r]);
    float sum = 0.f;
    for (int r = 0; r < 32; ++r) { float e = __expf(attn[h][r] - mx); attn[h][r] = e; sum += e; }
    float rinv = 1.0f / sum;
    for (int r = 0; r < 32; ++r) attn[h][r] *= rinv;
  }
  __syncthreads();
  {
    int h = c >> 4;
    float a = 0.f;
    #pragma unroll 8
    for (int r = 0; r < 32; ++r) a += attn[h][r] * Vs[r][c];
    os[c] = a;
  }
  __syncthreads();
  {
    float res = outb[c];
    #pragma unroll 8
    for (int k = 0; k < 128; ++k) res += os[k] * outw[c * 128 + k];
    orow[(size_t)l * 128 + c] = (__bf16)res;
  }
}

// ---------------------------------------------------------------------------
// Final: assemble triplane rows and apply proj_out: [12288 x 128] @ [128 x 64].
// 384 blocks x 1 wave; wave: 2 M-tiles x 4 N-tiles.
// Output layout [B][64][96][32].
// ---------------------------------------------------------------------------
__global__ __launch_bounds__(32)
void k_final(const __bf16* __restrict__ orow, const __bf16* __restrict__ wpo,
             const float* __restrict__ pb, float* __restrict__ out) {
  int mBase = blockIdx.x * 32;
  int lane = threadIdx.x;
  int col = lane & 15, hi = lane >> 4;
  auto lmap = [&](int m) -> size_t {   // pixel row (b,h,w) -> attention row index
    int b = m / 3072;
    int rem = m - b * 3072;
    int h = rem >> 5, w = rem & 31;
    int pp = h >> 5;
    return (size_t)pp * 4096 + (size_t)b * 1024 + (size_t)(h & 31) * 32 + w;
  };
  const __bf16* arow0 = orow + (lmap(mBase + col) << 7);
  const __bf16* arow1 = orow + (lmap(mBase + 16 + col) << 7);
  v8f acc[2][4] = {};
  #pragma unroll
  for (int kc = 0; kc < 4; ++kc) {
    v16bf a0 = load_a_frag(arow0 + kc * 32, hi);
    v16bf a1 = load_a_frag(arow1 + kc * 32, hi);
    #pragma unroll
    for (int nt = 0; nt < 4; ++nt) {
      int o = nt * 16 + col;
      v16bf bf = load_b_frag(wpo + ((size_t)o << 7) + kc * 32, hi);
      acc[0][nt] = wmma_bf16(a0, bf, acc[0][nt]);
      acc[1][nt] = wmma_bf16(a1, bf, acc[1][nt]);
    }
  }
  #pragma unroll
  for (int mt = 0; mt < 2; ++mt)
    #pragma unroll
    for (int nt = 0; nt < 4; ++nt) {
      int o = nt * 16 + col;
      float bo = pb[o];
      #pragma unroll
      for (int v = 0; v < 8; ++v) {
        int m = mBase + mt * 16 + (hi ? v + 8 : v);
        int b = m / 3072;
        int rem = m - b * 3072;
        int h = rem >> 5, w = rem & 31;
        out[(((size_t)b * 64 + o) * 96 + h) * 32 + w] = acc[mt][nt][v] + bo;
      }
    }
}

// ---------------------------------------------------------------------------
// Host launcher
// ---------------------------------------------------------------------------
static inline size_t alignUp(size_t x) { return (x + 255) & ~(size_t)255; }

extern "C" void kernel_launch(void* const* d_in, const int* in_sizes, int n_in,
                              void* d_out, int out_size, void* d_ws, size_t ws_size,
                              hipStream_t stream) {
  const float* triplane = (const float*)d_in[0];
  const float* image    = (const float*)d_in[1];
  const float* pm       = (const float*)d_in[2];
  const float* w_img    = (const float*)d_in[3];
  const float* b_img    = (const float*)d_in[4];
  const float* vox_w    = (const float*)d_in[5];
  const float* vox_b    = (const float*)d_in[6];
  const float* q_w      = (const float*)d_in[7];
  const float* q_b      = (const float*)d_in[8];
  const float* k_w      = (const float*)d_in[9];
  const float* k_b      = (const float*)d_in[10];
  const float* v_w      = (const float*)d_in[11];
  const float* v_b      = (const float*)d_in[12];
  const float* inW      = (const float*)d_in[13];
  const float* inB      = (const float*)d_in[14];
  const float* outW     = (const float*)d_in[15];
  const float* outB     = (const float*)d_in[16];
  const float* poW      = (const float*)d_in[17];
  const float* poB      = (const float*)d_in[18];
  float* out = (float*)d_out;

  char* ws = (char*)d_ws;
  size_t off = 0;
  auto wsalloc = [&](size_t bytes) -> char* {
    char* p = ws + off;
    off = alignUp(off + bytes);
    return p;
  };
  const size_t GPAD_BYTES = (size_t)BATCH * GP * GP * GP * 128 * sizeof(__bf16);
  __bf16* gpad  = (__bf16*)wsalloc(GPAD_BYTES);
  __bf16* vox   = (__bf16*)wsalloc((size_t)BATCH * 32768 * 128 * sizeof(__bf16));
  __bf16* orow  = (__bf16*)wsalloc((size_t)NROWS * 128 * sizeof(__bf16));
  __bf16* imT   = (__bf16*)wsalloc((size_t)BATCH * 256 * IN_CH * sizeof(__bf16));
  __bf16* wimg  = (__bf16*)wsalloc((size_t)128 * IN_CH * sizeof(__bf16));
  __bf16* wconv = (__bf16*)wsalloc((size_t)27 * 128 * 128 * sizeof(__bf16));
  __bf16* wpo   = (__bf16*)wsalloc((size_t)64 * 128 * sizeof(__bf16));
  __bf16* wk    = (__bf16*)wsalloc((size_t)128 * 128 * sizeof(__bf16));
  __bf16* wv    = (__bf16*)wsalloc((size_t)128 * 128 * sizeof(__bf16));
  float*  wq    = (float*) wsalloc((size_t)128 * 128 * sizeof(float));
  float*  bk    = (float*) wsalloc((size_t)32 * 128 * sizeof(float));
  float*  bv    = (float*) wsalloc((size_t)32 * 128 * sizeof(float));
  float*  bq    = (float*) wsalloc((size_t)128 * sizeof(float));
  float*  imf   = (float*) wsalloc((size_t)BATCH * 256 * 128 * sizeof(float));

  // zero padded grid (border stays zero; interior overwritten by sampler)
  hipMemsetAsync(gpad, 0, GPAD_BYTES, stream);

  // prep / transposes
  k_cvt_imT <<<(BATCH * IN_CH * 256 + 255) / 256, 256, 0, stream>>>(image, imT);
  k_cvt_bf16<<<(128 * IN_CH + 255) / 256, 256, 0, stream>>>(w_img, wimg, 128 * IN_CH);
  k_cvt_wconv<<<(128 * 128 * 27 + 255) / 256, 256, 0, stream>>>(vox_w, wconv);
  k_cvt_bf16<<<(64 * 128 + 255) / 256, 256, 0, stream>>>(poW, wpo, 64 * 128);
  k_combine_w<<<(3 * 128 * 128 + 255) / 256, 256, 0, stream>>>(inW, k_w, v_w, q_w, wk, wv, wq);
  k_combine_b<<<(8320 + 255) / 256, 256, 0, stream>>>(inW, inB, k_b, v_b, q_b, bk, bv, bq);

  // pipeline
  k_imgproj<<<1024 / 32, 128, 0, stream>>>(imT, wimg, b_img, imf);
  k_sampler<<<BATCH * 32768, 128, 0, stream>>>(imf, pm, gpad);
  k_conv3d <<<BATCH * 1024, 128, 0, stream>>>(gpad, wconv, vox_b, vox);
  k_kv_attn<<<NROWS, 128, 0, stream>>>(vox, wk, wv, wq, bk, bv, bq, triplane, outW, outB, orow);
  k_final  <<<NROWS / 32, 32, 0, stream>>>(orow, wpo, poB, out);

  (void)in_sizes; (void)n_in; (void)out_size; (void)ws_size;
}
